// Decoder_1374389534862
// MI455X (gfx1250) — matrix operand order repaired and checked by reading, not compiled, not run on hardware
//
#include <hip/hip_runtime.h>
#include <hip/hip_bf16.h>
#include <cmath>

#define B_  512
#define Zl  64
#define Hd  256
#define Td  64
#define Pd  10000
#define G3  768   // 3*H

typedef __attribute__((ext_vector_type(16))) _Float16 v16h;
typedef __attribute__((ext_vector_type(8)))  _Float16 v8h;
typedef __attribute__((ext_vector_type(8)))  float    v8f;

__device__ __forceinline__ float sigmoidf_(float x){ return 1.0f/(1.0f + __expf(-x)); }
__device__ __forceinline__ float eluf_(float x){ return x > 0.0f ? x : (__expf(x)-1.0f); }

// ---------------- small prep kernels ----------------

__global__ void k_h0(const float* __restrict__ z, const float* __restrict__ Wi,
                     const float* __restrict__ bi, float* __restrict__ h0f){
  int idx = blockIdx.x*blockDim.x + threadIdx.x;
  if (idx >= B_*Hd) return;
  int b = idx >> 8, j = idx & 255;
  float acc = bi[j];
  const float* zr = z + b*Zl;
  const float* wr = Wi + j*Zl;
  #pragma unroll 8
  for (int k=0;k<Zl;k++) acc += zr[k]*wr[k];
  h0f[idx] = eluf_(acc);
}

// gx0 = W_ih0 @ emb + b_ih0 + b_hh0  (constant over b,t since x == embedding)
__global__ void k_gx0(const float* __restrict__ Wih0, const float* __restrict__ emb,
                      const float* __restrict__ bih0, const float* __restrict__ bhh0,
                      float* __restrict__ gx0){
  int g = blockIdx.x*blockDim.x + threadIdx.x;
  if (g >= G3) return;
  float acc = bih0[g] + bhh0[g];
  const float* wr = Wih0 + g*Hd;
  #pragma unroll 8
  for (int k=0;k<Hd;k++) acc += wr[k]*emb[k];
  gx0[g] = acc;
}

__global__ void k_cvt(const float* __restrict__ s, _Float16* __restrict__ d, int n){
  int i = blockIdx.x*blockDim.x + threadIdx.x;
  if (i < n) d[i] = (_Float16)s[i];
}

// ---------------- GRU layer 0 (constant input gates) ----------------
// One block of 256 threads (8 waves) owns 16 batch rows for all T steps.
// LDS: gh f32 [16][776] | h32 f32 [16][256] | h16 f16 [16][264] | gxs f32 [768]
#define GHS 776
#define HS16 264

__global__ void __launch_bounds__(256) k_gru0(
    const _Float16* __restrict__ Whh, const float* __restrict__ gx0,
    const float* __restrict__ h0f, _Float16* __restrict__ out0){
  extern __shared__ char smem[];
  float*    gh  = (float*)smem;                 // 49664 B
  float*    h32 = (float*)(smem + 49664);       // 16384 B
  _Float16* h16 = (_Float16*)(smem + 66048);    //  8448 B
  float*    gxs = (float*)(smem + 74496);       //  3072 B

  const int tid = threadIdx.x;
  const int b0  = blockIdx.x * 16;

  { // init hidden state + gate constants
    int m = tid >> 4, j0 = (tid & 15) * 16;
    for (int jj=0;jj<16;jj++){
      float v = h0f[(b0+m)*Hd + j0 + jj];
      h32[m*Hd + j0+jj]  = v;
      h16[m*HS16 + j0+jj] = (_Float16)v;
    }
    for (int g = tid; g < G3; g += 256) gxs[g] = gx0[g];
  }
  __syncthreads();

  const int wave  = tid >> 5, l = tid & 31;
  const int mA    = l & 15;
  const int aoff  = (l >> 4) * 16;    // A fragment k-offset within 32-k block
  const int nlane = l & 15;
  const int boff  = (l >> 4) * 16;    // B fragment k-offset
  const int mr    = (l >> 4) * 8;     // C fragment row offset

  for (int t=0; t<Td; t++){
    v8f acc[6] = {};
    #pragma unroll
    for (int kb=0; kb<8; kb++){
      const _Float16* ap = h16 + mA*HS16 + kb*32 + aoff;
      union { v16h v; v8h h[2]; } ua;
      ua.h[0] = *(const v8h*)ap;
      ua.h[1] = *(const v8h*)(ap + 8);
      #pragma unroll
      for (int i=0;i<6;i++){
        int n = (wave*6 + i)*16 + nlane;
        v16h bv = *(const v16h*)(Whh + n*Hd + kb*32 + boff);
        acc[i] = __builtin_amdgcn_wmma_f32_16x16x32_f16(
                   false, ua.v, false, bv, (short)0, acc[i], false, false);
      }
    }
    #pragma unroll
    for (int i=0;i<6;i++){
      int n = (wave*6 + i)*16 + nlane;
      #pragma unroll
      for (int r=0;r<8;r++) gh[(mr+r)*GHS + n] = acc[i][r];
    }
    __syncthreads();

    { // gate math
      int m = tid >> 4, j0 = (tid & 15)*16;
      for (int jj=0;jj<16;jj++){
        int j = j0 + jj;
        float r  = sigmoidf_(gxs[j]        + gh[m*GHS + j]);
        float u  = sigmoidf_(gxs[Hd + j]   + gh[m*GHS + Hd + j]);
        float nn = tanhf   (gxs[2*Hd + j]  + r*gh[m*GHS + 2*Hd + j]);
        float hn = (1.0f-u)*nn + u*h32[m*Hd + j];
        h32[m*Hd + j]   = hn;
        h16[m*HS16 + j] = (_Float16)hn;
        out0[((size_t)(b0+m)*Td + t)*Hd + j] = (_Float16)hn;
      }
    }
    __syncthreads();
  }
}

// ---------------- GEMM: gx1 = out0 @ W_ih1^T + (b_ih1 + b_hh1), f16 out ----------------
__global__ void __launch_bounds__(256) k_gemm_gx1(
    const _Float16* __restrict__ A,   // (32768, 256)
    const _Float16* __restrict__ W,   // (768, 256)
    const float* __restrict__ bih, const float* __restrict__ bhh,
    _Float16* __restrict__ C){        // (32768, 768)
  const int tid = threadIdx.x;
  const int wave = tid >> 5, l = tid & 31;
  const int m0 = blockIdx.y*128 + wave*16;
  const int nb = blockIdx.x*64;
  const int mA = m0 + (l & 15);
  const int aoff = (l >> 4)*16;
  const int nlane = l & 15, boff = (l >> 4)*16;
  v8f acc[4] = {};
  #pragma unroll
  for (int kb=0; kb<8; kb++){
    const _Float16* ap = A + (size_t)mA*Hd + kb*32 + aoff;
    union { v16h v; v8h h[2]; } ua;
    ua.h[0] = *(const v8h*)ap;
    ua.h[1] = *(const v8h*)(ap + 8);
    #pragma unroll
    for (int i=0;i<4;i++){
      int n = nb + i*16 + nlane;
      v16h bv = *(const v16h*)(W + (size_t)n*Hd + kb*32 + boff);
      acc[i] = __builtin_amdgcn_wmma_f32_16x16x32_f16(
                 false, ua.v, false, bv, (short)0, acc[i], false, false);
    }
  }
  const int mr = (l >> 4)*8;
  #pragma unroll
  for (int i=0;i<4;i++){
    int n = nb + i*16 + nlane;
    float bias = bih[n] + bhh[n];
    #pragma unroll
    for (int r=0;r<8;r++){
      int m = m0 + mr + r;
      C[(size_t)m*G3 + n] = (_Float16)(acc[i][r] + bias);
    }
  }
}

// ---------------- GRU layer 1 + fused LayerNorm + ELU ----------------
__global__ void __launch_bounds__(256) k_gru1(
    const _Float16* __restrict__ Whh, const _Float16* __restrict__ gx1,
    const float* __restrict__ h0f, const float* __restrict__ lng,
    const float* __restrict__ lnb, _Float16* __restrict__ yout){
  extern __shared__ char smem[];
  float*    gh   = (float*)smem;                 // 49664
  float*    h32  = (float*)(smem + 49664);       // 16384
  _Float16* h16  = (_Float16*)(smem + 66048);    //  8448
  float*    sg   = (float*)(smem + 74496);       //  1024
  float*    sb   = (float*)(smem + 75520);       //  1024
  float*    reds = (float*)(smem + 76544);       //  1024
  float*    redq = (float*)(smem + 77568);       //  1024

  const int tid = threadIdx.x;
  const int b0  = blockIdx.x * 16;

  {
    int m = tid >> 4, j0 = (tid & 15) * 16;
    for (int jj=0;jj<16;jj++){
      float v = h0f[(b0+m)*Hd + j0 + jj];
      h32[m*Hd + j0+jj]  = v;
      h16[m*HS16 + j0+jj] = (_Float16)v;
    }
    sg[tid] = lng[tid];
    sb[tid] = lnb[tid];
  }
  __syncthreads();

  const int wave  = tid >> 5, l = tid & 31;
  const int mAf   = l & 15;
  const int aoff  = (l >> 4) * 16;
  const int nlane = l & 15;
  const int boff  = (l >> 4) * 16;
  const int mr    = (l >> 4) * 8;

  for (int t=0; t<Td; t++){
    v8f acc[6] = {};
    #pragma unroll
    for (int kb=0; kb<8; kb++){
      const _Float16* ap = h16 + mAf*HS16 + kb*32 + aoff;
      union { v16h v; v8h h[2]; } ua;
      ua.h[0] = *(const v8h*)ap;
      ua.h[1] = *(const v8h*)(ap + 8);
      #pragma unroll
      for (int i=0;i<6;i++){
        int n = (wave*6 + i)*16 + nlane;
        v16h bv = *(const v16h*)(Whh + n*Hd + kb*32 + boff);
        acc[i] = __builtin_amdgcn_wmma_f32_16x16x32_f16(
                   false, ua.v, false, bv, (short)0, acc[i], false, false);
      }
    }
    #pragma unroll
    for (int i=0;i<6;i++){
      int n = (wave*6 + i)*16 + nlane;
      #pragma unroll
      for (int r=0;r<8;r++) gh[(mr+r)*GHS + n] = acc[i][r];
    }
    __syncthreads();

    int m = tid >> 4, g = tid & 15, j0 = g*16;
    const _Float16* gxp = gx1 + ((size_t)(b0+m)*Td + t)*G3;
    float hv16[16];
    float ls = 0.0f, lq = 0.0f;
    for (int jj=0;jj<16;jj++){
      int j = j0 + jj;
      float r  = sigmoidf_((float)gxp[j]        + gh[m*GHS + j]);
      float u  = sigmoidf_((float)gxp[Hd + j]   + gh[m*GHS + Hd + j]);
      float nn = tanhf   ((float)gxp[2*Hd + j]  + r*gh[m*GHS + 2*Hd + j]);
      float hn = (1.0f-u)*nn + u*h32[m*Hd + j];
      h32[m*Hd + j]   = hn;
      h16[m*HS16 + j] = (_Float16)hn;
      hv16[jj] = hn; ls += hn; lq += hn*hn;
    }
    reds[m*16+g] = ls; redq[m*16+g] = lq;
    __syncthreads();

    float mu = 0.0f, qq = 0.0f;
    for (int g2=0; g2<16; g2++){ mu += reds[m*16+g2]; qq += redq[m*16+g2]; }
    mu *= (1.0f/256.0f); qq *= (1.0f/256.0f);
    float rs = rsqrtf(qq - mu*mu + 1e-5f);
    for (int jj=0;jj<16;jj++){
      int j = j0 + jj;
      float yv = (hv16[jj]-mu)*rs*sg[j] + sb[j];
      yout[((size_t)(b0+m)*Td + t)*Hd + j] = (_Float16)eluf_(yv);
    }
    __syncthreads();
  }
}

// ---------------- GEMM: logits = y @ W_out^T + b_out, f32 out ----------------
__global__ void __launch_bounds__(256) k_gemm_out(
    const _Float16* __restrict__ A,   // (32768, 256)
    const _Float16* __restrict__ W,   // (10000, 256)
    const float* __restrict__ bout,
    float* __restrict__ C){           // (32768, 10000)
  const int tid = threadIdx.x;
  const int wave = tid >> 5, l = tid & 31;
  const int m0 = blockIdx.y*128 + wave*16;
  const int nb = blockIdx.x*64;
  const int mA = m0 + (l & 15);
  const int aoff = (l >> 4)*16;
  const int nlane = l & 15, boff = (l >> 4)*16;
  v8f acc[4] = {};
  #pragma unroll
  for (int kb=0; kb<8; kb++){
    const _Float16* ap = A + (size_t)mA*Hd + kb*32 + aoff;
    union { v16h v; v8h h[2]; } ua;
    ua.h[0] = *(const v8h*)ap;
    ua.h[1] = *(const v8h*)(ap + 8);
    #pragma unroll
    for (int i=0;i<4;i++){
      int n  = nb + i*16 + nlane;
      int nr = n < Pd ? n : (Pd-1);           // clamp (no divergence, EXEC stays full)
      v16h bv = *(const v16h*)(W + (size_t)nr*Hd + kb*32 + boff);
      acc[i] = __builtin_amdgcn_wmma_f32_16x16x32_f16(
                 false, ua.v, false, bv, (short)0, acc[i], false, false);
    }
  }
  const int mr = (l >> 4)*8;
  #pragma unroll
  for (int i=0;i<4;i++){
    int n = nb + i*16 + nlane;
    if (n < Pd){
      float bias = bout[n];
      #pragma unroll
      for (int r=0;r<8;r++){
        int m = m0 + mr + r;
        C[(size_t)m*Pd + n] = acc[i][r] + bias;
      }
    }
  }
}

// ---------------- launcher ----------------

extern "C" void kernel_launch(void* const* d_in, const int* in_sizes, int n_in,
                              void* d_out, int out_size, void* d_ws, size_t ws_size,
                              hipStream_t stream){
  const float* z      = (const float*)d_in[0];
  const float* W_init = (const float*)d_in[1];
  const float* b_init = (const float*)d_in[2];
  const float* emb    = (const float*)d_in[3];
  const float* W_ih0  = (const float*)d_in[4];
  const float* W_hh0  = (const float*)d_in[5];
  const float* b_ih0  = (const float*)d_in[6];
  const float* b_hh0  = (const float*)d_in[7];
  const float* W_ih1  = (const float*)d_in[8];
  const float* W_hh1  = (const float*)d_in[9];
  const float* b_ih1  = (const float*)d_in[10];
  const float* b_hh1  = (const float*)d_in[11];
  const float* ln_g   = (const float*)d_in[12];
  const float* ln_b   = (const float*)d_in[13];
  const float* W_out  = (const float*)d_in[14];
  const float* b_out  = (const float*)d_in[15];
  float* out = (float*)d_out;

  char* ws = (char*)d_ws;
  size_t off = 0;
  auto carve = [&](size_t bytes)->char*{
    char* p = ws + off; off = (off + bytes + 255) & ~(size_t)255; return p; };

  float*    h0f   = (float*)   carve((size_t)B_*Hd*4);
  float*    gx0   = (float*)   carve((size_t)G3*4);
  _Float16* Whh0h = (_Float16*)carve((size_t)G3*Hd*2);
  _Float16* Wih1h = (_Float16*)carve((size_t)G3*Hd*2);
  _Float16* Whh1h = (_Float16*)carve((size_t)G3*Hd*2);
  _Float16* Wouth = (_Float16*)carve((size_t)Pd*Hd*2);
  _Float16* out0h = (_Float16*)carve((size_t)B_*Td*Hd*2);
  _Float16* gx1h  = (_Float16*)carve((size_t)B_*Td*G3*2);
  _Float16* yh    = (_Float16*)carve((size_t)B_*Td*Hd*2);
  (void)ws_size; (void)in_sizes; (void)n_in; (void)out_size;

  k_h0 <<<(B_*Hd+255)/256, 256, 0, stream>>>(z, W_init, b_init, h0f);
  k_gx0<<<(G3+255)/256,    256, 0, stream>>>(W_ih0, emb, b_ih0, b_hh0, gx0);
  k_cvt<<<(G3*Hd+255)/256, 256, 0, stream>>>(W_hh0, Whh0h, G3*Hd);
  k_cvt<<<(G3*Hd+255)/256, 256, 0, stream>>>(W_ih1, Wih1h, G3*Hd);
  k_cvt<<<(G3*Hd+255)/256, 256, 0, stream>>>(W_hh1, Whh1h, G3*Hd);
  k_cvt<<<(Pd*Hd+255)/256, 256, 0, stream>>>(W_out, Wouth, Pd*Hd);

  k_gru0<<<B_/16, 256, 77568, stream>>>(Whh0h, gx0, h0f, out0h);

  k_gemm_gx1<<<dim3(G3/64, (B_*Td)/128), 256, 0, stream>>>(out0h, Wih1h, b_ih1, b_hh1, gx1h);

  k_gru1<<<B_/16, 256, 78592, stream>>>(Whh1h, gx1h, h0f, ln_g, ln_b, yh);

  k_gemm_out<<<dim3((Pd+63)/64, (B_*Td)/128), 256, 0, stream>>>(yh, Wouth, b_out, out);
}
